// RecursiveAttnPooling_42941083025461
// MI455X (gfx1250) — matrix operand order repaired and checked by reading, not compiled, not run on hardware
//
#include <hip/hip_runtime.h>
#include <hip/hip_bf16.h>

// ---------------- problem constants ----------------
#define BB 8
#define TT 1500
#define DD 1536
#define DP 1024
#define EE 192
#define ITERS 6
#define NEG_INF -1.0e30f

typedef __attribute__((ext_vector_type(16))) __bf16 v16bf;
typedef __attribute__((ext_vector_type(8)))  float  v8f;

union Frag16 { v16bf v; uint4 q[2]; };

__device__ __forceinline__ unsigned short f2bf(float x) {
    unsigned int u = __float_as_uint(x);
    unsigned int r = u + 0x7FFFu + ((u >> 16) & 1u);   // round-to-nearest-even
    return (unsigned short)(r >> 16);
}

__device__ __forceinline__ float wave_sum(float v) {
    #pragma unroll
    for (int o = 16; o > 0; o >>= 1) v += __shfl_xor(v, o, 32);
    return v;
}

// ---------------- init ----------------
__global__ void k_init(float* __restrict__ C, const float* __restrict__ C0,
                       int* __restrict__ stopArr, float* __restrict__ maskArr) {
    int idx = blockIdx.x * blockDim.x + threadIdx.x;
    if (idx < BB * DD) C[idx] = C0[idx % DD];
    if (idx < BB * ITERS) maskArr[idx] = 1.0f;
    if (idx < BB) stopArr[idx] = 0;
}

// ---------------- fp32 -> bf16 convert (x4 vectorized) ----------------
__global__ void k_f2bf4(ushort4* __restrict__ dst, const float4* __restrict__ src, int n4) {
    int idx = blockIdx.x * blockDim.x + threadIdx.x;
    if (idx >= n4) return;
    float4 v = src[idx];
    ushort4 o;
    o.x = f2bf(v.x); o.y = f2bf(v.y); o.z = f2bf(v.z); o.w = f2bf(v.w);
    dst[idx] = o;
}

// ------- pack row-major KxN fp32 weight into WMMA B-fragment layout (bf16) -------
// fragment layout: tile (kt, nt): lane l element j <- W[kt*32 + (l>=16?8:0) + (j&7) + (j>=8?16:0), nt*16 + (l&15)]
__global__ void k_packB(unsigned short* __restrict__ dst, const float* __restrict__ src,
                        int K, int N) {
    int Ntiles = N / 16;
    int n = (K / 32) * Ntiles * 512;
    int idx = blockIdx.x * blockDim.x + threadIdx.x;
    if (idx >= n) return;
    int j    = idx & 15;
    int l    = (idx >> 4) & 31;
    int tile = idx >> 9;
    int kt = tile / Ntiles, nt = tile % Ntiles;
    int row = kt * 32 + ((l >= 16) ? 8 : 0) + (j & 7) + ((j >= 8) ? 16 : 0);
    int col = nt * 16 + (l & 15);
    dst[idx] = f2bf(src[(size_t)row * N + col]);
}

// ---------------- uniform stats (mu0, sigma0) ----------------
__global__ void k_stats0(const float* __restrict__ h, float* __restrict__ mu0,
                         float* __restrict__ sigma0) {
    int idx = blockIdx.x * blockDim.x + threadIdx.x;
    if (idx >= BB * DD) return;
    int b = idx / DD, d = idx % DD;
    float s = 0.f, s2 = 0.f;
    const float* p = h + (size_t)b * TT * DD + d;
    for (int t = 0; t < TT; ++t) { float x = p[(size_t)t * DD]; s += x; s2 += x * x; }
    float inv = 1.0f / (float)TT;
    float mu = s * inv;
    mu0[idx] = mu;
    float v = s2 * inv - mu * mu;
    sigma0[idx] = sqrtf(fmaxf(v, 1e-8f));
}

// ---------------- energy mask (wave per row) ----------------
__global__ void k_energy(const float* __restrict__ h, float* __restrict__ energy) {
    int w = (blockIdx.x * (blockDim.x >> 5)) + (threadIdx.x >> 5);
    int lane = threadIdx.x & 31;
    if (w >= BB * TT) return;
    const float* p = h + (size_t)w * DD;
    float s = 0.f;
    for (int d = lane; d < DD; d += 32) { float x = p[d]; s += x * x; }
    s = wave_sum(s);
    if (lane == 0) energy[w] = ((s / (float)DD) > 1e-4f) ? 1.0f : 0.0f;
}

// ---------------- per-batch bias from mu0/sigma0 through W1 rows [D:3D) ----------------
__global__ void k_bias(const float* __restrict__ mu0, const float* __restrict__ sigma0,
                       const float* __restrict__ W1, float* __restrict__ bias) {
    int idx = blockIdx.x * blockDim.x + threadIdx.x;
    if (idx >= BB * DP) return;
    int b = idx / DP, p = idx % DP;
    float acc = 0.f;
    const float* mu = mu0 + b * DD;
    const float* sg = sigma0 + b * DD;
    for (int d = 0; d < DD; ++d) {
        acc += mu[d] * W1[(size_t)(DD + d) * DP + p];
        acc += sg[d] * W1[(size_t)(2 * DD + d) * DP + p];
    }
    bias[idx] = acc;
}

// ---------------- WMMA GEMM: Out[M,N] = A(bf16,row-major MxK) @ Bp(packed) + bias ----------------
// Each wave: 32(M) x 64(N) tile, 8 accumulators, register double-buffered K pipeline.
// MODE 0: bias = rowBias[(row/TT)*N + col]   MODE 1: bias = colBias[col]
template <int K, int N, int MODE>
__global__ void __launch_bounds__(128)
k_wmma_gemm(const unsigned short* __restrict__ A, const unsigned short* __restrict__ Bp,
            const float* __restrict__ rowBias, const float* __restrict__ colBias,
            float* __restrict__ Out, int M) {
    constexpr int Ktiles = K / 32;
    constexpr int BSTR   = (N / 16) * 1024;     // bytes per K-step in packed B
    constexpr int A2OFF  = 32 * K;              // bytes to second 16-row M tile

    int wid  = blockIdx.x * (blockDim.x >> 5) + (threadIdx.x >> 5);
    int lane = threadIdx.x & 31;
    int nGroups = N / 64;
    int mt = wid / nGroups;
    int ng = wid % nGroups;
    if (mt * 32 >= M) return;
    int m0 = mt * 32, n0 = ng * 64;

    v8f acc[8];
    #pragma unroll
    for (int i = 0; i < 8; ++i) acc[i] = (v8f){0.f,0.f,0.f,0.f,0.f,0.f,0.f,0.f};

    int aOff = (lane < 16) ? 0 : 8;
    const char* pa = (const char*)(A + (size_t)(m0 + (lane & 15)) * K + aOff);
    const char* pb = (const char*)(Bp + ((size_t)(n0 >> 4) * 32 + lane) * 16);

    Frag16 a0[2], b0[4], a1[2], b1[4];

    auto LD = [&](Frag16* a, Frag16* b, const char* qa, const char* qb) {
        a[0].q[0] = *(const uint4*)(qa);
        a[0].q[1] = *(const uint4*)(qa + 32);
        a[1].q[0] = *(const uint4*)(qa + A2OFF);
        a[1].q[1] = *(const uint4*)(qa + A2OFF + 32);
        #pragma unroll
        for (int f = 0; f < 4; ++f) {
            b[f].q[0] = *(const uint4*)(qb + f * 1024);
            b[f].q[1] = *(const uint4*)(qb + f * 1024 + 16);
        }
    };
    auto MM = [&](Frag16* a, Frag16* b) {
        #pragma unroll
        for (int mi = 0; mi < 2; ++mi)
            #pragma unroll
            for (int f = 0; f < 4; ++f)
                acc[mi * 4 + f] = __builtin_amdgcn_wmma_f32_16x16x32_bf16(
                    false, a[mi].v, false, b[f].v, (short)0, acc[mi * 4 + f], false, false);
    };

    LD(a0, b0, pa, pb); pa += 64; pb += BSTR;
    #pragma unroll 1
    for (int kt = 0; kt + 2 <= Ktiles; kt += 2) {
        LD(a1, b1, pa, pb); pa += 64; pb += BSTR;      // prefetch kt+1
        MM(a0, b0);
        if (kt + 2 < Ktiles) { LD(a0, b0, pa, pb); pa += 64; pb += BSTR; }  // prefetch kt+2
        MM(a1, b1);
    }

    int colLane = lane & 15;
    int rowHi = (lane < 16) ? 0 : 8;
    #pragma unroll
    for (int mi = 0; mi < 2; ++mi) {
        #pragma unroll
        for (int f = 0; f < 4; ++f) {
            int col = n0 + f * 16 + colLane;
            #pragma unroll
            for (int v = 0; v < 8; ++v) {
                int row = m0 + mi * 16 + v + rowHi;
                float bv = (MODE == 0) ? rowBias[(size_t)(row / TT) * N + col] : colBias[col];
                Out[(size_t)row * N + col] = acc[mi * 4 + f][v] + bv;
            }
        }
    }
}

// ---------------- C @ Wc  (B x Dp) ----------------
__global__ void k_cwc(const float* __restrict__ C, const float* __restrict__ Wc,
                      float* __restrict__ CWc) {
    int idx = blockIdx.x * blockDim.x + threadIdx.x;
    if (idx >= BB * DP) return;
    int b = idx / DP, p = idx % DP;
    float acc = 0.f;
    const float* c = C + b * DD;
    for (int d = 0; d < DD; ++d) acc += c[d] * Wc[(size_t)d * DP + p];
    CWc[idx] = acc;
}

// ---------------- relu(base + CWc) -> bf16 (x4 vectorized) ----------------
__global__ void k_act4(const float4* __restrict__ base, const float4* __restrict__ CWc,
                       ushort4* __restrict__ act) {
    int idx = blockIdx.x * blockDim.x + threadIdx.x;
    if (idx >= (BB * TT * DP) / 4) return;
    int b  = idx / ((TT * DP) / 4);
    int p4 = idx % (DP / 4);
    float4 v = base[idx];
    float4 c = CWc[b * (DP / 4) + p4];
    ushort4 o;
    o.x = f2bf(fmaxf(v.x + c.x, 0.f));
    o.y = f2bf(fmaxf(v.y + c.y, 0.f));
    o.z = f2bf(fmaxf(v.z + c.z, 0.f));
    o.w = f2bf(fmaxf(v.w + c.w, 0.f));
    act[idx] = o;
}

// ---------------- att_raw = masked mean over D (wave per row) ----------------
__global__ void k_att(const float* __restrict__ out2, const float* __restrict__ energy,
                      float* __restrict__ att) {
    int w = (blockIdx.x * (blockDim.x >> 5)) + (threadIdx.x >> 5);
    int lane = threadIdx.x & 31;
    if (w >= BB * TT) return;
    const float* p = out2 + (size_t)w * DD;
    float s = 0.f;
    for (int d = lane; d < DD; d += 32) s += p[d];
    s = wave_sum(s);
    if (lane == 0) att[w] = (energy[w] > 0.5f) ? (s / (float)DD) : NEG_INF;
}

// ---------------- softmax over T, then renormalize by (sum+1e-8) ----------------
__global__ void k_softmax(const float* __restrict__ att, float* __restrict__ A) {
    int b = blockIdx.x, tid = threadIdx.x;
    __shared__ float red[256];
    const float* x = att + b * TT;
    float m = -3.0e38f;
    for (int t = tid; t < TT; t += 256) m = fmaxf(m, x[t]);
    red[tid] = m; __syncthreads();
    for (int s = 128; s > 0; s >>= 1) { if (tid < s) red[tid] = fmaxf(red[tid], red[tid + s]); __syncthreads(); }
    m = red[0]; __syncthreads();
    float s = 0.f;
    for (int t = tid; t < TT; t += 256) s += expf(x[t] - m);
    red[tid] = s; __syncthreads();
    for (int st = 128; st > 0; st >>= 1) { if (tid < st) red[tid] += red[tid + st]; __syncthreads(); }
    float S = red[0]; __syncthreads();
    float inv = 1.0f / S;
    float s2 = 0.f;
    for (int t = tid; t < TT; t += 256) { float a = expf(x[t] - m) * inv; A[b * TT + t] = a; s2 += a; }
    red[tid] = s2; __syncthreads();
    for (int st = 128; st > 0; st >>= 1) { if (tid < st) red[tid] += red[tid + st]; __syncthreads(); }
    float inv2 = 1.0f / (red[0] + 1e-8f);
    for (int t = tid; t < TT; t += 256) A[b * TT + t] *= inv2;
}

// ---------------- per-(b,d) reductions over T ----------------
__global__ void k_statsP(const float* __restrict__ h, const float* __restrict__ out2,
                         const float* __restrict__ A, float* __restrict__ mu_p,
                         float* __restrict__ sigma_p, float* __restrict__ Ah,
                         float* __restrict__ Ao) {
    int idx = blockIdx.x * blockDim.x + threadIdx.x;
    if (idx >= BB * DD) return;
    int b = idx / DD, d = idx % DD;
    const float* ph = h    + (size_t)b * TT * DD + d;
    const float* po = out2 + (size_t)b * TT * DD + d;
    const float* pa = A + b * TT;
    float mu = 0.f, sec = 0.f, ah = 0.f, ao = 0.f;
    for (int t = 0; t < TT; ++t) {
        float x = ph[(size_t)t * DD];
        float o = po[(size_t)t * DD];
        float a = pa[t];
        mu += o * x; sec += o * x * x; ah += a * x; ao += a * o;
    }
    mu_p[idx] = mu;
    sigma_p[idx] = sqrtf(fmaxf(sec - mu * mu, 1e-8f));
    Ah[idx] = ah;
    Ao[idx] = ao;
}

// ---------------- emb raw = concat(mu_p,sigma_p) @ w0 + b0 ----------------
__global__ void k_emb(const float* __restrict__ mu_p, const float* __restrict__ sigma_p,
                      const float* __restrict__ w0, const float* __restrict__ b0,
                      float* __restrict__ embRaw) {
    int idx = blockIdx.x * blockDim.x + threadIdx.x;
    if (idx >= BB * EE) return;
    int b = idx / EE, e = idx % EE;
    float acc = b0[e];
    const float* mu = mu_p + b * DD;
    const float* sg = sigma_p + b * DD;
    for (int d = 0; d < DD; ++d) {
        acc += mu[d] * w0[(size_t)d * EE + e];
        acc += sg[d] * w0[(size_t)(DD + d) * EE + e];
    }
    embRaw[idx] = acc;
}

// ---------------- head: C update, stop prob, state, normalized emb out ----------------
__global__ void k_head(const float* __restrict__ embRaw, const float* __restrict__ Ah,
                       const float* __restrict__ Ao, float* __restrict__ C,
                       int* __restrict__ stopArr, float* __restrict__ maskArr,
                       const float* __restrict__ Wstop, const float* __restrict__ bstop,
                       const float* __restrict__ threshold,
                       float* __restrict__ outE, float* __restrict__ outP, int iter) {
    int w = threadIdx.x >> 5;   // wave = batch
    int lane = threadIdx.x & 31;
    if (w >= BB) return;
    int b = w;
    int wasStop = stopArr[b];
    float scale = wasStop ? 0.0f : (1.0f / (float)TT);
    float dot = 0.f;
    for (int d = lane; d < DD; d += 32) {
        float cv = C[b * DD + d] + scale * Ah[b * DD + d];
        C[b * DD + d] = cv;                      // wave b owns row b
        dot += Ao[b * DD + d] * Wstop[d];
        dot += cv * Wstop[DD + d];
    }
    dot = wave_sum(dot);
    float ssq = 0.f;
    for (int e = lane; e < EE; e += 32) { float v = embRaw[b * EE + e]; ssq += v * v; }
    ssq = wave_sum(ssq);
    float denom = fmaxf(sqrtf(ssq), 1e-12f);
    for (int e = lane; e < EE; e += 32)
        outE[(size_t)(b * ITERS + iter) * EE + e] = embRaw[b * EE + e] / denom;
    if (lane == 0) {
        float p = 1.0f / (1.0f + expf(-(dot + bstop[0])));
        outP[b * ITERS + iter] = p;
        int newly = (p < threshold[0]) ? 1 : 0;
        if (newly) for (int k = iter; k < ITERS; ++k) maskArr[b * ITERS + k] = 0.0f;
        stopArr[b] = wasStop | newly;
    }
}

// ---------------- final mask application on E ----------------
__global__ void k_mask(float* __restrict__ outE, const float* __restrict__ maskArr) {
    int idx = blockIdx.x * blockDim.x + threadIdx.x;
    if (idx >= BB * ITERS * EE) return;
    int bk = idx / EE;   // b*ITERS + k
    outE[idx] *= maskArr[bk];
}

// ================= launch =================
extern "C" void kernel_launch(void* const* d_in, const int* in_sizes, int n_in,
                              void* d_out, int out_size, void* d_ws, size_t ws_size,
                              hipStream_t stream) {
    const float* h    = (const float*)d_in[0];
    const float* W1   = (const float*)d_in[1];
    const float* Wc   = (const float*)d_in[2];
    const float* W2   = (const float*)d_in[3];
    const float* b2   = (const float*)d_in[4];
    const float* w0   = (const float*)d_in[5];
    const float* b0   = (const float*)d_in[6];
    const float* Wst  = (const float*)d_in[7];
    const float* bst  = (const float*)d_in[8];
    const float* thr  = (const float*)d_in[9];
    const float* C0   = (const float*)d_in[10];

    float* outE = (float*)d_out;                   // [B,6,E]
    float* outP = outE + BB * ITERS * EE;          // [B,6]

    const size_t NH = (size_t)BB * TT * DD;        // 18,432,000
    const size_t NB = (size_t)BB * TT * DP;        // 12,288,000

    char* base_p = (char*)d_ws;
    size_t off = 0;
    auto alloc = [&](size_t bytes) -> char* {
        char* p = base_p + off;
        off = (off + bytes + 255) & ~(size_t)255;
        return p;
    };
    unsigned short* hbf  = (unsigned short*)alloc(NH * 2);
    unsigned short* w1p  = (unsigned short*)alloc((size_t)DD * DP * 2);
    unsigned short* w2p  = (unsigned short*)alloc((size_t)DP * DD * 2);
    float* baseBuf       = (float*)alloc(NB * 4);
    unsigned short* act  = (unsigned short*)alloc(NB * 2);
    float* out2          = (float*)alloc(NH * 4);
    float* mu0    = (float*)alloc(BB * DD * 4);
    float* sigma0 = (float*)alloc(BB * DD * 4);
    float* biasBp = (float*)alloc(BB * DP * 4);
    float* energy = (float*)alloc(BB * TT * 4);
    float* CWc    = (float*)alloc(BB * DP * 4);
    float* att    = (float*)alloc(BB * TT * 4);
    float* Aarr   = (float*)alloc(BB * TT * 4);
    float* mu_p   = (float*)alloc(BB * DD * 4);
    float* sig_p  = (float*)alloc(BB * DD * 4);
    float* Ah     = (float*)alloc(BB * DD * 4);
    float* Ao     = (float*)alloc(BB * DD * 4);
    float* embRaw = (float*)alloc(BB * EE * 4);
    float* Cst    = (float*)alloc(BB * DD * 4);
    int*   stopA  = (int*)alloc(BB * 4);
    float* maskA  = (float*)alloc(BB * ITERS * 4);
    if (off > ws_size) return;   // insufficient workspace

    const int TB = 256;
    // init + conversions + packing
    k_init<<<(BB * DD + TB - 1) / TB, TB, 0, stream>>>(Cst, C0, stopA, maskA);
    k_f2bf4<<<(int)((NH / 4 + TB - 1) / TB), TB, 0, stream>>>((ushort4*)hbf, (const float4*)h, (int)(NH / 4));
    k_packB<<<((DD / 32) * (DP / 16) * 512 + TB - 1) / TB, TB, 0, stream>>>(w1p, W1, DD, DP);
    k_packB<<<((DP / 32) * (DD / 16) * 512 + TB - 1) / TB, TB, 0, stream>>>(w2p, W2, DP, DD);
    k_stats0<<<(BB * DD + TB - 1) / TB, TB, 0, stream>>>(h, mu0, sigma0);
    k_energy<<<(BB * TT + 3) / 4, 128, 0, stream>>>(h, energy);
    k_bias<<<(BB * DP + TB - 1) / TB, TB, 0, stream>>>(mu0, sigma0, W1, biasBp);

    // GEMM1: base = h_bf16 @ W1h + bias[b, :]   (32x64 per wave)
    {
        int waves = ((BB * TT) / 32) * (DP / 64);   // 375 * 16 = 6000
        k_wmma_gemm<DD, DP, 0><<<waves / 4, 128, 0, stream>>>(hbf, w1p, biasBp, b2,
                                                              baseBuf, BB * TT);
    }

    for (int it = 0; it < ITERS; ++it) {
        k_cwc<<<(BB * DP + TB - 1) / TB, TB, 0, stream>>>(Cst, Wc, CWc);
        k_act4<<<(int)((NB / 4 + TB - 1) / TB), TB, 0, stream>>>((const float4*)baseBuf,
                                                                 (const float4*)CWc,
                                                                 (ushort4*)act);
        {
            int waves = ((BB * TT) / 32) * (DD / 64);  // 375 * 24 = 9000
            k_wmma_gemm<DP, DD, 1><<<waves / 4, 128, 0, stream>>>(act, w2p, biasBp, b2,
                                                                  out2, BB * TT);
        }
        k_att<<<(BB * TT + 3) / 4, 128, 0, stream>>>(out2, energy, att);
        k_softmax<<<BB, 256, 0, stream>>>(att, Aarr);
        k_statsP<<<(BB * DD + TB - 1) / TB, TB, 0, stream>>>(h, out2, Aarr,
                                                             mu_p, sig_p, Ah, Ao);
        k_emb<<<(BB * EE + TB - 1) / TB, TB, 0, stream>>>(mu_p, sig_p, w0, b0, embRaw);
        k_head<<<1, 256, 0, stream>>>(embRaw, Ah, Ao, Cst, stopA, maskA,
                                      Wst, bst, thr, outE, outP, it);
    }
    k_mask<<<(BB * ITERS * EE + TB - 1) / TB, TB, 0, stream>>>(outE, maskA);
}
